// Model_22050362097798
// MI455X (gfx1250) — matrix operand order repaired and checked by reading, not compiled, not run on hardware
//
#include <hip/hip_runtime.h>
#include <math.h>

// ---------------- WMMA types ----------------
typedef __attribute__((ext_vector_type(16))) __bf16 v16bf;
typedef __attribute__((ext_vector_type(8)))  float  v8f;

#define HID   512
#define HID3  1536
#define VOCAB 100
#define VPAD  112
#define BATCH 2048
#define TMAX  201

// round-to-nearest-even float -> bf16 bits (stored as ushort)
__device__ __forceinline__ unsigned short f2bf(float x) {
    unsigned int u = __float_as_uint(x);
    u += 0x7FFFu + ((u >> 16) & 1u);
    return (unsigned short)(u >> 16);
}

// A fragment: 16x32 bf16, lane holds row m=lane&15, g=lane>>4.
// VGPR pair p: K = 16g + 2p => 16 contiguous bf16 = two b128 loads.
__device__ __forceinline__ v16bf load_fragA(const unsigned short* p, int row, int kbase, int g) {
    union { uint4 q[2]; v16bf v; } u;
    const unsigned short* base = p + row * HID + kbase;
    u.q[0] = *(const uint4*)(base + 16 * g);
    u.q[1] = *(const uint4*)(base + 16 * g + 8);
    return u.v;
}

// B fragment: 32x16 bf16 from row-major W[n, k] (B[k][n] = W[n][k]).
// Lane holds col n=lane&15; VGPR pair p: K = 16g + 2p => 16 contiguous bf16 = two b128 loads.
__device__ __forceinline__ v16bf load_fragB(const unsigned short* p, int nrow, int kbase, int g) {
    union { uint4 q[2]; v16bf v; } u;
    const uint4* q = (const uint4*)(p + nrow * HID + kbase + 16 * g);
    u.q[0] = q[0];
    u.q[1] = q[1];
    return u.v;
}

__device__ __forceinline__ v8f wmma_bf16(v16bf a, v16bf b, v8f c) {
    return __builtin_amdgcn_wmma_f32_16x16x32_bf16(false, a, false, b, (short)0, c, false, false);
}

// ---------------- one-time prep kernels ----------------
__global__ void k_convert_whh(const float* __restrict__ whh, unsigned short* __restrict__ out) {
    int i = blockIdx.x * 256 + threadIdx.x;           // 1536*512
    out[i] = f2bf(whh[i]);
}

__global__ void k_convert_wp(const float* __restrict__ wp, unsigned short* __restrict__ out) {
    int i = blockIdx.x * 256 + threadIdx.x;           // 112*512
    int row = i / HID;
    out[i] = (row < VOCAB) ? f2bf(wp[i]) : (unsigned short)0;
}

// E_ih[v, j] = b_ih[j] + sum_k embed[v,k] * W_ih[j,k]   (turns the x-GEMM into a gather)
__global__ void k_eih(const float* __restrict__ embed, const float* __restrict__ wih,
                      const float* __restrict__ bih, float* __restrict__ eih) {
    int idx = blockIdx.x * 128 + threadIdx.x;         // 100*1536
    int v = idx / HID3, j = idx % HID3;
    const float* e = embed + v * HID;
    const float* w = wih + j * HID;
    float s = bih[j];
    for (int k = 0; k < HID; ++k) s = fmaf(e[k], w[k], s);
    eih[idx] = s;
}

__global__ void k_init(const float* __restrict__ feat, const int* __restrict__ sos,
                       float* __restrict__ h, unsigned short* __restrict__ hb,
                       int* __restrict__ pred) {
    int idx = blockIdx.x * 256 + threadIdx.x;         // 2048*512
    float v = feat[idx];
    h[idx] = v;
    hb[idx] = f2bf(v);
    if (idx < BATCH) pred[idx] = sos[0];
}

// ---------------- per-step kernels ----------------
// Fused GRU step: for hidden columns [j0, j0+64), compute all 3 gate GEMM slices
// (N-cols j0 / 512+j0 / 1024+j0) so r/z/n for a given (row, col) land in matching
// accumulator slots; then do the elementwise GRU combine in-register and write
// the new h (fp32 + bf16) directly. No [2048,1536] scratch round-trip.
// Block = 128 thr (4 waves); block tile = 64(M) x 64(h-cols); wave = 16(M) x 12 WMMA tiles.
// Ping-pong buffers: reads h_in/hb_in, writes h_out/hb_out (blocks sharing M-rows
// across blockIdx.y read the full K range, so in-place would race).
__global__ void k_gru(const unsigned short* __restrict__ hb_in,
                      const float* __restrict__ h_in,
                      const unsigned short* __restrict__ whh,
                      const float* __restrict__ eih,
                      const float* __restrict__ bhh,
                      const int* __restrict__ pred,
                      float* __restrict__ h_out,
                      unsigned short* __restrict__ hb_out) {
    int lane = threadIdx.x & 31;
    int wave = threadIdx.x >> 5;
    int g = lane >> 4;
    int ln = lane & 15;
    int mbase = blockIdx.x * 64 + wave * 16;
    int j0 = blockIdx.y * 64;
    int mrow = mbase + ln;

    v8f acc[3][4] = {};
    for (int k = 0; k < HID; k += 32) {
        v16bf a = load_fragA(hb_in, mrow, k, g);
#pragma unroll
        for (int gate = 0; gate < 3; ++gate)
#pragma unroll
            for (int c = 0; c < 4; ++c) {
                v16bf b = load_fragB(whh, gate * HID + j0 + c * 16 + ln, k, g);
                acc[gate][c] = wmma_bf16(a, b, acc[gate][c]);
            }
    }

    // in-register GRU combine: same (c, r, lane) slot across the 3 gate accs
#pragma unroll
    for (int c = 0; c < 4; ++c) {
        int j = j0 + c * 16 + ln;
        float br = bhh[j];
        float bz = bhh[HID + j];
        float bn = bhh[2 * HID + j];
#pragma unroll
        for (int r = 0; r < 8; ++r) {
            int row = mbase + r + 8 * g;
            const float* e = eih + (size_t)pred[row] * HID3;   // gi = E_ih[pred] gather
            float gr = acc[0][c][r] + br + e[j];
            float gz = acc[1][c][r] + bz + e[HID + j];
            float ghn = acc[2][c][r] + bn;
            float rr = 1.0f / (1.0f + __expf(-gr));
            float zz = 1.0f / (1.0f + __expf(-gz));
            float nn = tanhf(e[2 * HID + j] + rr * ghn);
            float hn = (1.0f - zz) * nn + zz * h_in[(size_t)row * HID + j];
            h_out[(size_t)row * HID + j] = hn;
            hb_out[(size_t)row * HID + j] = f2bf(hn);
        }
    }
}

// logits = h @ W_proj^T + b_proj ; greedy argmax -> pred ; store out[b, v, t].
// Block = 128 thr (4 waves) per 16-row batch tile; VOCAB padded to 112 = 7 N-tiles.
__global__ void k_logits(const unsigned short* __restrict__ hb,
                         const unsigned short* __restrict__ wp,
                         const float* __restrict__ bproj,
                         float* __restrict__ out, int* __restrict__ pred, int t) {
    __shared__ float lds[16 * VPAD];
    int lane = threadIdx.x & 31;
    int wave = threadIdx.x >> 5;
    int g = lane >> 4;
    int ln = lane & 15;
    int mbase = blockIdx.x * 16;
    int mrow = mbase + ln;
    int n0 = wave * 16;
    int n1 = (wave + 4) * 16;                          // valid only for wave < 3

    v8f acc0 = {}, acc1 = {};
    for (int k = 0; k < HID; k += 32) {
        v16bf a = load_fragA(hb, mrow, k, g);
        v16bf b0 = load_fragB(wp, n0 + ln, k, g);
        acc0 = wmma_bf16(a, b0, acc0);
        if (wave < 3) {                                // wave-uniform branch, EXEC stays all-1
            v16bf b1 = load_fragB(wp, n1 + ln, k, g);
            acc1 = wmma_bf16(a, b1, acc1);
        }
    }
#pragma unroll
    for (int r = 0; r < 8; ++r) {
        lds[(r + 8 * g) * VPAD + n0 + ln] = acc0[r];
        if (wave < 3) lds[(r + 8 * g) * VPAD + n1 + ln] = acc1[r];
    }
    __syncthreads();

    if (threadIdx.x < 16) {                            // per-row greedy argmax (first max wins)
        int row = threadIdx.x;
        float best = -INFINITY;
        int bi = 0;
        for (int v = 0; v < VOCAB; ++v) {
            float val = lds[row * VPAD + v] + bproj[v];
            if (val > best) { best = val; bi = v; }
        }
        pred[mbase + row] = bi;
    }
    // out layout: [B, V, T]
    for (int idx = threadIdx.x; idx < 16 * VOCAB; idx += 128) {
        int row = idx / VOCAB, v = idx % VOCAB;
        out[(size_t)(mbase + row) * (VOCAB * TMAX) + (size_t)v * TMAX + t] =
            lds[row * VPAD + v] + bproj[v];
    }
}

// ---------------- host launcher ----------------
extern "C" void kernel_launch(void* const* d_in, const int* in_sizes, int n_in,
                              void* d_out, int out_size, void* d_ws, size_t ws_size,
                              hipStream_t stream) {
    const float* feat   = (const float*)d_in[0];
    const float* W_ih   = (const float*)d_in[1];
    const float* W_hh   = (const float*)d_in[2];
    const float* b_ih   = (const float*)d_in[3];
    const float* b_hh   = (const float*)d_in[4];
    const float* W_proj = (const float*)d_in[5];
    const float* b_proj = (const float*)d_in[6];
    const float* embed  = (const float*)d_in[7];
    const int*   sos    = (const int*)d_in[8];
    float* out = (float*)d_out;

    char* ws = (char*)d_ws;
    unsigned short* whh_b = (unsigned short*)ws;  ws += (size_t)HID3 * HID * 2;   // 1.5 MB
    unsigned short* wp_b  = (unsigned short*)ws;  ws += (size_t)VPAD * HID * 2;   // 112 KB
    float*          eih   = (float*)ws;           ws += (size_t)VOCAB * HID3 * 4; // 600 KB
    float*          h0    = (float*)ws;           ws += (size_t)BATCH * HID * 4;  // 4 MB
    float*          h1    = (float*)ws;           ws += (size_t)BATCH * HID * 4;  // 4 MB
    unsigned short* hb0   = (unsigned short*)ws;  ws += (size_t)BATCH * HID * 2;  // 2 MB
    unsigned short* hb1   = (unsigned short*)ws;  ws += (size_t)BATCH * HID * 2;  // 2 MB
    int*            pred  = (int*)ws;                                             // 8 KB

    // one-time prep (inside the captured graph; trivial cost)
    k_convert_whh<<<(HID3 * HID) / 256, 256, 0, stream>>>(W_hh, whh_b);
    k_convert_wp<<<(VPAD * HID) / 256, 256, 0, stream>>>(W_proj, wp_b);
    k_eih<<<(VOCAB * HID3) / 128, 128, 0, stream>>>(embed, W_ih, b_ih, eih);
    k_init<<<(BATCH * HID) / 256, 256, 0, stream>>>(feat, sos, h0, hb0, pred);

    float*          h_in  = h0;  float*          h_out = h1;
    unsigned short* hb_in = hb0; unsigned short* hb_out = hb1;
    for (int t = 0; t < TMAX; ++t) {
        k_gru<<<dim3(BATCH / 64, HID / 64), 128, 0, stream>>>(
            hb_in, h_in, whh_b, eih, b_hh, pred, h_out, hb_out);
        k_logits<<<BATCH / 16, 128, 0, stream>>>(hb_out, wp_b, b_proj, out, pred, t);
        // ping-pong
        float* th = h_in;           h_in = h_out;   h_out = th;
        unsigned short* tb = hb_in; hb_in = hb_out; hb_out = tb;
    }
}
